// MiniGRUConv2d4_3650722201912
// MI455X (gfx1250) — compile-verified, matrix operand order checked by reading, not run on hardware
//
#include <hip/hip_runtime.h>
#include <hip/hip_bf16.h>

typedef __attribute__((ext_vector_type(16))) _Float16 v16h;
typedef __attribute__((ext_vector_type(4)))  _Float16 v4h;
typedef __attribute__((ext_vector_type(8)))  float    v8f;
typedef __attribute__((ext_vector_type(4)))  int      v4i;

// Geometry (fixed by the reference problem)
#define BSZ   4
#define CIN   64
#define CGRP  64
#define C4    256
#define HH    256
#define WW    256
#define KTOT  576           // 64 * 9
#define KSNUM 18            // K steps of 32

// ---- CDNA5 async VMEM->LDS path, guarded so compile never breaks ----------
#if defined(__HIP_DEVICE_COMPILE__) && defined(__gfx1250__) && \
    __has_builtin(__builtin_amdgcn_global_load_async_to_lds_b128) && \
    __has_builtin(__builtin_amdgcn_s_wait_asynccnt)
#define USE_ASYNC_LDS 1
#else
#define USE_ASYNC_LDS 0
#endif

#if USE_ASYNC_LDS
typedef __attribute__((address_space(1))) v4i* gptr_v4i;   // global int4*
typedef __attribute__((address_space(3))) v4i* lptr_v4i;   // LDS int4*
#endif

static __device__ __forceinline__ void cp16_lds(_Float16* dst, const _Float16* src) {
#if USE_ASYNC_LDS
    __builtin_amdgcn_global_load_async_to_lds_b128(
        (gptr_v4i)(void*)src, (lptr_v4i)(void*)dst, 0, 0);
#else
    *(uint4*)dst = *(const uint4*)src;
#endif
}
static __device__ __forceinline__ void stage_fence() {
#if USE_ASYNC_LDS
    __builtin_amdgcn_s_wait_asynccnt(0);
#endif
}

// -------------------------------------------------------------------------
// Pre-pack 1: xs (B,Cin,H,W) f32 -> xh NHWC f16 (channel fastest).
// -------------------------------------------------------------------------
__global__ __launch_bounds__(256) void pack_x_f16(
    const float* __restrict__ xs, _Float16* __restrict__ xh)
{
    const int t   = blockIdx.x * blockDim.x + threadIdx.x;   // 4,194,304
    const int ci4 = (t & 15) * 4;
    const int pix = t >> 4;                                  // 0..262143
    const int b   = pix >> 16;
    const int off = pix & 65535;
    v4h hv;
    #pragma unroll
    for (int j = 0; j < 4; ++j)
        hv[j] = (_Float16)xs[(((size_t)b * CIN + ci4 + j) << 16) + off];
    *(v4h*)&xh[(size_t)pix * 64 + ci4] = hv;
}

// -------------------------------------------------------------------------
// Pre-pack 2: weights -> f16 in per-K-step LDS layout:
//   wpk[((t*18 + ks)*256 + cc)*32 + kk] = w_t[cc][ci0+kk][tap]
// -------------------------------------------------------------------------
__global__ __launch_bounds__(256) void pack_w_f16(
    const float* __restrict__ wz, const float* __restrict__ wh,
    const float* __restrict__ wss, _Float16* __restrict__ wpk)
{
    const int t     = blockIdx.x * blockDim.x + threadIdx.x;  // 442,368
    const int kk    = t & 31;
    const int cc    = (t >> 5) & 255;
    const int plane = t >> 13;                                // 0..53
    const int tt    = plane / 18;
    const int ks    = plane - tt * 18;
    const int tap   = ks >> 1;
    const int ci    = (ks & 1) * 32 + kk;
    const float* W3[3] = {wz, wh, wss};
    wpk[t] = (_Float16)W3[tt][(size_t)cc * KTOT + ci * 9 + tap];
}

// -------------------------------------------------------------------------
// Kernel 1: implicit-GEMM 3x3 conv + fused BN affine (+sigmoid for z,s).
// M tile = 64 contiguous x-pixels of one (b,y) row; N tile = 128 channels.
// 8 waves as 2(M) x 4(N); each wave owns 2x2 WMMA accumulators.
// Fully-unrolled 18 K-steps; staging is async VMEM->LDS b128 copies when
// the gfx1250 builtins are exposed, else register copies.
// -------------------------------------------------------------------------
__global__ __launch_bounds__(256, 2) void conv_bn_wmma(
    const _Float16* __restrict__ xh, const _Float16* __restrict__ wpk,
    const float* __restrict__ gz, const float* __restrict__ bz, const float* __restrict__ mz, const float* __restrict__ vz,
    const float* __restrict__ gh, const float* __restrict__ bh, const float* __restrict__ mh, const float* __restrict__ vh,
    const float* __restrict__ gs, const float* __restrict__ bs, const float* __restrict__ ms, const float* __restrict__ vs,
    float* __restrict__ zbuf, float* __restrict__ hbuf, float* __restrict__ sbuf)
{
    __shared__ __align__(32) _Float16 Alds[64 * 32];    // [m][k]
    __shared__ __align__(32) _Float16 Blds[128 * 32];   // [n][k]

    const int tid  = threadIdx.x;
    const int lane = tid & 31;
    const int wv   = tid >> 5;
    const int wm   = wv & 1;           // 2 wave-rows (M)
    const int wn   = wv >> 1;          // 4 wave-cols (N)
    const int lr   = lane & 15;
    const int kh   = lane >> 4;

    const int mblk = blockIdx.x;       // 4096
    const int rid  = mblk >> 2;
    const int x0   = (mblk & 3) * 64;
    const int bb   = rid >> 8;
    const int y    = rid & 255;

    const int nb   = blockIdx.y * 128; // 6 N-blocks over 768
    const int tb   = nb >> 8;          // 0=z,1=h,2=s
    const int c0   = nb & 255;

    const float* G3[3]  = {gz, gh, gs};
    const float* Bi3[3] = {bz, bh, bs};
    const float* M3[3]  = {mz, mh, ms};
    const float* V3[3]  = {vz, vh, vs};
    float*       O3[3]  = {zbuf, hbuf, sbuf};

    // Staging coordinates (one 16B A copy, two 16B B copies per thread/step)
    const int am  = tid >> 2;          // pixel within tile
    const int akg = (tid & 3) * 8;     // channel-octet within 32-k block
    const int bnn = tid >> 1;          // output channel within 128-tile
    const int bkg = (tid & 1) * 16;    // k-16-group
    _Float16* adst = &Alds[am * 32 + akg];
    _Float16* bdst = &Blds[bnn * 32 + bkg];
    const ptrdiff_t apixBase =
        (((ptrdiff_t)(bb << 16) + (y << 8) + (x0 + am)) << 6) + akg;
    const _Float16* bsrcBase =
        wpk + (((size_t)tb * KSNUM * 256) + (c0 + bnn)) * 32 + bkg;

    // Fragment LDS pointers (invariant across K-steps)
    const _Float16* ap0 = &Alds[(wm * 32 + lr) * 32 + kh * 16];
    const _Float16* ap1 = ap0 + 16 * 32;
    const _Float16* bp0 = &Blds[(wn * 32 + lr) * 32 + kh * 16];
    const _Float16* bp1 = bp0 + 16 * 32;

    v8f acc[2][2] = {};

    #pragma unroll
    for (int tap = 0; tap < 9; ++tap) {
        const int dy = tap / 3 - 1;            // compile-time after unroll
        const int dx = tap % 3 - 1;
        const int yy = y + dy;
        const int xx = x0 + am + dx;
        const bool inb = ((unsigned)yy < (unsigned)HH) &&
                         ((unsigned)xx < (unsigned)WW);
        #pragma unroll
        for (int half = 0; half < 2; ++half) {
            const int ks  = tap * 2 + half;
            const int ci0 = half * 32;
            // ---- stage A: one 16B copy (zero-fill at the halo) ----
            if (inb) {
                const _Float16* asrc =
                    xh + apixBase + (ptrdiff_t)dy * (256 * 64) +
                    (ptrdiff_t)dx * 64 + ci0;
                cp16_lds(adst, asrc);
            } else {
                *(uint4*)adst = make_uint4(0u, 0u, 0u, 0u);
            }
            // ---- stage B: two 16B copies from the pre-packed layout ----
            const _Float16* bsrc = bsrcBase + (size_t)ks * 8192;
            cp16_lds(bdst, bsrc);
            cp16_lds(bdst + 8, bsrc + 8);

            stage_fence();
            __syncthreads();

            const v16h a0 = *(const v16h*)ap0;
            const v16h a1 = *(const v16h*)ap1;
            const v16h b0 = *(const v16h*)bp0;
            const v16h b1 = *(const v16h*)bp1;
            acc[0][0] = __builtin_amdgcn_wmma_f32_16x16x32_f16(
                false, a0, false, b0, (short)0, acc[0][0], false, false);
            acc[0][1] = __builtin_amdgcn_wmma_f32_16x16x32_f16(
                false, a0, false, b1, (short)0, acc[0][1], false, false);
            acc[1][0] = __builtin_amdgcn_wmma_f32_16x16x32_f16(
                false, a1, false, b0, (short)0, acc[1][0], false, false);
            acc[1][1] = __builtin_amdgcn_wmma_f32_16x16x32_f16(
                false, a1, false, b1, (short)0, acc[1][1], false, false);
            __syncthreads();
        }
    }

    // ---- epilogue: BN affine (+sigmoid for z,s) ----
    float* obuf = O3[tb];
    #pragma unroll
    for (int mt = 0; mt < 2; ++mt) {
        #pragma unroll
        for (int nt = 0; nt < 2; ++nt) {
            const int nl = wn * 32 + nt * 16 + lr;
            const int cc = c0 + nl;
            const float inv  = G3[tb][cc] * __frsqrt_rn(V3[tb][cc] + 1e-5f);
            const float bias = Bi3[tb][cc] - M3[tb][cc] * inv;
            #pragma unroll
            for (int r = 0; r < 8; ++r) {
                const int ml = wm * 32 + mt * 16 + r + 8 * kh;
                float val = acc[mt][nt][r] * inv + bias;
                if (tb != 1) val = 1.0f / (1.0f + __expf(-val));
                obuf[(((size_t)bb * C4 + cc) << 16) + (y << 8) + (x0 + ml)] = val;
            }
        }
    }
}

// -------------------------------------------------------------------------
// Kernel 2: four directional minGRU scans, in-place on hbuf.
// -------------------------------------------------------------------------
__global__ __launch_bounds__(256) void scan4(
    const float* __restrict__ zbuf, float* __restrict__ hbuf,
    const float* __restrict__ h20, const float* __restrict__ h21,
    const float* __restrict__ h30, const float* __restrict__ h31)
{
    const int t  = blockIdx.x * blockDim.x + threadIdx.x;   // 262144
    const int g  = t >> 16;
    const int r  = t & 65535;
    const int b  = r >> 14;
    const int c  = (r >> 8) & 63;
    const int q  = r & 255;                 // x for g<2, y for g>=2
    const int ch = g * CGRP + c;
    const size_t plane = ((size_t)b * C4 + ch) << 16;

    if (g == 0) {
        float carry = h20[c];
        for (int yv = 0; yv < HH; ++yv) {
            const size_t i = plane + ((size_t)yv << 8) + q;
            if (yv + 4 < HH) { __builtin_prefetch(&zbuf[i + 1024], 0, 1);
                               __builtin_prefetch(&hbuf[i + 1024], 1, 1); }
            const float z = zbuf[i], h = hbuf[i];
            carry = z * h + (1.0f - z) * carry;
            hbuf[i] = carry;
        }
    } else if (g == 1) {
        float carry = h21[c];
        for (int yv = HH - 1; yv >= 0; --yv) {
            const size_t i = plane + ((size_t)yv << 8) + q;
            const float z = zbuf[i], h = hbuf[i];
            carry = z * h + (1.0f - z) * carry;
            hbuf[i] = carry;
        }
    } else if (g == 2) {
        const size_t base = plane + ((size_t)q << 8);
        float carry = h30[c];
        for (int xv = 0; xv < WW; ++xv) {
            const size_t i = base + xv;
            const float z = zbuf[i], h = hbuf[i];
            carry = z * h + (1.0f - z) * carry;
            hbuf[i] = carry;
        }
    } else {
        const size_t base = plane + ((size_t)q << 8);
        float carry = h31[c];
        for (int xv = WW - 1; xv >= 0; --xv) {
            const size_t i = base + xv;
            const float z = zbuf[i], h = hbuf[i];
            carry = z * h + (1.0f - z) * carry;
            hbuf[i] = carry;
        }
    }
}

// -------------------------------------------------------------------------
// Kernel 3: out = sum_g s[g] * h_scanned[g]   (float4 vectorized)
// -------------------------------------------------------------------------
__global__ __launch_bounds__(256) void combine4(
    const float* __restrict__ sbuf, const float* __restrict__ hbuf,
    float* __restrict__ out)
{
    const int t  = blockIdx.x * blockDim.x + threadIdx.x;   // 4194304
    const int xq = t & 63;
    const int y  = (t >> 6) & 255;
    const int c  = (t >> 14) & 63;
    const int b  = t >> 20;
    const size_t inoff  = (((size_t)b * C4 + c) << 16) + ((size_t)y << 8) + (size_t)xq * 4;
    const size_t outoff = (((size_t)b * CGRP + c) << 16) + ((size_t)y << 8) + (size_t)xq * 4;

    float4 accv = make_float4(0.f, 0.f, 0.f, 0.f);
    #pragma unroll
    for (int g = 0; g < 4; ++g) {
        const size_t o = inoff + ((size_t)(g * CGRP) << 16);
        const float4 sv = *(const float4*)&sbuf[o];
        const float4 hv = *(const float4*)&hbuf[o];
        accv.x += sv.x * hv.x;  accv.y += sv.y * hv.y;
        accv.z += sv.z * hv.z;  accv.w += sv.w * hv.w;
    }
    *(float4*)&out[outoff] = accv;
}

// -------------------------------------------------------------------------
extern "C" void kernel_launch(void* const* d_in, const int* in_sizes, int n_in,
                              void* d_out, int out_size, void* d_ws, size_t ws_size,
                              hipStream_t stream) {
    const float* xs  = (const float*)d_in[0];
    const float* wz  = (const float*)d_in[1];
    const float* gz  = (const float*)d_in[2];
    const float* bz  = (const float*)d_in[3];
    const float* mz  = (const float*)d_in[4];
    const float* vz  = (const float*)d_in[5];
    const float* wh  = (const float*)d_in[6];
    const float* gh  = (const float*)d_in[7];
    const float* bh  = (const float*)d_in[8];
    const float* mh  = (const float*)d_in[9];
    const float* vh  = (const float*)d_in[10];
    const float* wss = (const float*)d_in[11];
    const float* gs  = (const float*)d_in[12];
    const float* bs  = (const float*)d_in[13];
    const float* ms  = (const float*)d_in[14];
    const float* vs  = (const float*)d_in[15];
    const float* h20 = (const float*)d_in[16];
    const float* h21 = (const float*)d_in[17];
    const float* h30 = (const float*)d_in[18];
    const float* h31 = (const float*)d_in[19];

    const size_t tensor_elems = (size_t)BSZ * C4 * HH * WW;   // 67,108,864 floats
    float* zbuf = (float*)d_ws;
    float* hbuf = zbuf + tensor_elems;
    float* sbuf = hbuf + tensor_elems;
    _Float16* xh  = (_Float16*)(sbuf + tensor_elems);         // 16,777,216 halves
    _Float16* wpk = xh + (size_t)BSZ * HH * WW * CIN;         // 442,368 halves

    pack_x_f16<<<16384, 256, 0, stream>>>(xs, xh);
    pack_w_f16<<<1728, 256, 0, stream>>>(wz, wh, wss, wpk);

    conv_bn_wmma<<<dim3(4096, 6), 256, 0, stream>>>(
        xh, wpk,
        gz, bz, mz, vz, gh, bh, mh, vh, gs, bs, ms, vs,
        zbuf, hbuf, sbuf);

    scan4<<<1024, 256, 0, stream>>>(zbuf, hbuf, h20, h21, h30, h31);
    combine4<<<16384, 256, 0, stream>>>(sbuf, hbuf, (float*)d_out);
}
